// MAB_32908039422414
// MI455X (gfx1250) — matrix operand order repaired and checked by reading, not compiled, not run on hardware
//
#include <hip/hip_runtime.h>
#include <hip/hip_bf16.h>

static constexpr int BB     = 4;
static constexpr int NN     = 2048;
static constexpr int DMODEL = 256;
static constexpr int HH     = 8;
static constexpr int HDD    = 32;
static constexpr int DFF    = 1024;

typedef __attribute__((ext_vector_type(16))) _Float16 v16h;
typedef __attribute__((ext_vector_type(8)))  float    v8f;

union V16H  { v16h v; _Float16 e[16]; uint4 q[2]; };
union V32H  { _Float16 e[32]; uint4 q[4]; };

__device__ inline v8f wmma_f16(v16h a, v16h b, v8f c) {
  // D = A(16x32 f16) * B(32x16 f16) + C(16x16 f32)
  return __builtin_amdgcn_wmma_f32_16x16x32_f16(false, a, false, b, (short)0, c, false, false);
}

// A fragment (16x32, f16) from a row-major [rows][pitch] matrix.
// Lane layout: row = r0 + lane%16; element e -> K = (e/8)*16 + (lane/16)*8 + e%8.
__device__ inline v16h load_a_frag(const _Float16* base, int pitch, int r0, int k0, int lane) {
  V16H f;
  const int row  = r0 + (lane & 15);
  const int ksub = (lane >> 4) * 8;
  f.q[0] = *(const uint4*)(base + (size_t)row * pitch + k0 + ksub);
  f.q[1] = *(const uint4*)(base + (size_t)row * pitch + k0 + 16 + ksub);
  return f.v;
}

// B fragment (32x16, f16) from N-major storage baseNK[n][k] (pitch = K stride).
// Lane layout: col = n0 + lane%16; element e -> K = k0 + (e/8)*16 + (lane/16)*8 + e%8.
__device__ inline v16h load_b_frag_nk(const _Float16* baseNK, int pitch, int n0, int k0, int lane) {
  V16H f;
  const int col = n0 + (lane & 15);
  const int kk  = k0 + (lane >> 4) * 8;
  f.q[0] = *(const uint4*)(baseNK + (size_t)col * pitch + kk);
  f.q[1] = *(const uint4*)(baseNK + (size_t)col * pitch + kk + 16);
  return f.v;
}

// ---------------- LayerNorm: fp32 row -> f16 normed row (1 wave / row) -------
__global__ __launch_bounds__(32) void ln_to_f16(const float* __restrict__ x,
                                                const float* __restrict__ gamma,
                                                const float* __restrict__ beta,
                                                _Float16* __restrict__ out) {
  const int row  = blockIdx.x;
  const int lane = threadIdx.x;
  const float* xr = x + (size_t)row * DMODEL;
  float vals[8];
  {
    float4 a0 = ((const float4*)xr)[lane];
    float4 a1 = ((const float4*)xr)[lane + 32];
    vals[0] = a0.x; vals[1] = a0.y; vals[2] = a0.z; vals[3] = a0.w;
    vals[4] = a1.x; vals[5] = a1.y; vals[6] = a1.z; vals[7] = a1.w;
  }
  float s = 0.0f;
#pragma unroll
  for (int j = 0; j < 8; ++j) s += vals[j];
#pragma unroll
  for (int off = 1; off < 32; off <<= 1) s += __shfl_xor(s, off, 32);
  const float mu = s * (1.0f / DMODEL);
  float vv = 0.0f;
#pragma unroll
  for (int j = 0; j < 8; ++j) { const float d = vals[j] - mu; vv += d * d; }
#pragma unroll
  for (int off = 1; off < 32; off <<= 1) vv += __shfl_xor(vv, off, 32);
  const float rs = rsqrtf(vv * (1.0f / DMODEL) + 1e-5f);
#pragma unroll
  for (int j = 0; j < 4; ++j) {
    const int idx = lane * 4 + j;
    out[(size_t)row * DMODEL + idx] = (_Float16)((vals[j] - mu) * rs * gamma[idx] + beta[idx]);
  }
#pragma unroll
  for (int j = 0; j < 4; ++j) {
    const int idx = (lane + 32) * 4 + j;
    out[(size_t)row * DMODEL + idx] = (_Float16)((vals[4 + j] - mu) * rs * gamma[idx] + beta[idx]);
  }
}

// ---------------- Flash attention with adjacency mask ------------------------
// 1 wave per (b, h, 16-query tile). Online softmax; S and O tiles via WMMA.
__global__ __launch_bounds__(32) void attn_kernel(const float* __restrict__ x,
                                                  const int* __restrict__ adj,
                                                  const _Float16* __restrict__ xn,
                                                  float* __restrict__ x1) {
  constexpr int KP = 40;  // LDS row pitch in halves (80B: 16B-aligned, bank-spread)
  __shared__ __align__(16) _Float16 kbuf [32 * KP];  // [key][hd]   (B operand for S)
  __shared__ __align__(16) _Float16 kbufT[32 * KP];  // [hd][key]   (B operand for O)
  __shared__ __align__(16) _Float16 pbuf [16 * KP];  // [qrow][key] (A operand for O)

  const int lane = threadIdx.x;
  const int q0   = blockIdx.x * 16;
  const int h    = blockIdx.y;
  const int b    = blockIdx.z;

  const _Float16* xb = xn + (size_t)b * NN * DMODEL;
  const v16h qf = load_a_frag(xb, DMODEL, q0, h * HDD, lane);

  v8f o0 = {}; v8f o1 = {};
  float m[8], l[8];
#pragma unroll
  for (int r = 0; r < 8; ++r) { m[r] = -__builtin_inff(); l[r] = 0.0f; }

  const int   rowb  = (lane >> 4) * 8;
  const int   colb  = lane & 15;
  const float scale = 0.17677669529663687f;  // 1/sqrt(32)

  for (int kt = 0; kt < NN; kt += 32) {
    // Stage 32 keys x 32 dims into LDS, both [key][hd] and transposed [hd][key].
    {
      V32H row;
      const uint4* src = (const uint4*)(xb + (size_t)(kt + lane) * DMODEL + h * HDD);
      row.q[0] = src[0]; row.q[1] = src[1]; row.q[2] = src[2]; row.q[3] = src[3];
      uint4* dst = (uint4*)(kbuf + lane * KP);
      dst[0] = row.q[0]; dst[1] = row.q[1]; dst[2] = row.q[2]; dst[3] = row.q[3];
#pragma unroll
      for (int j = 0; j < 32; ++j) kbufT[j * KP + lane] = row.e[j];
      if (kt + 32 < NN) {
        __builtin_prefetch(xb + (size_t)(kt + 32 + lane) * DMODEL + h * HDD, 0, 3);
      }
    }
    __syncthreads();

    const v16h bk0 = load_b_frag_nk(kbuf, KP, 0, 0, lane);
    const v16h bk1 = load_b_frag_nk(kbuf, KP, 16, 0, lane);
    v8f s0 = {}; v8f s1 = {};
    s0 = wmma_f16(qf, bk0, s0);
    s1 = wmma_f16(qf, bk1, s1);

#pragma unroll
    for (int r = 0; r < 8; ++r) {
      const int  qrow = q0 + rowb + r;
      const int* arow = adj + ((size_t)b * NN + qrow) * NN + kt;
      const bool v0 = arow[colb] > 0;
      const bool v1 = arow[16 + colb] > 0;
      const float sv0 = v0 ? s0[r] * scale : -__builtin_inff();
      const float sv1 = v1 ? s1[r] * scale : -__builtin_inff();
      float t = fmaxf(sv0, sv1);
      t = fmaxf(t, __shfl_xor(t, 1, 32));
      t = fmaxf(t, __shfl_xor(t, 2, 32));
      t = fmaxf(t, __shfl_xor(t, 4, 32));
      t = fmaxf(t, __shfl_xor(t, 8, 32));
      const float mnew  = fmaxf(m[r], t);
      const float alpha = (mnew == -__builtin_inff()) ? 1.0f : expf(m[r] - mnew);
      const float p0 = v0 ? expf(sv0 - mnew) : 0.0f;
      const float p1 = v1 ? expf(sv1 - mnew) : 0.0f;
      l[r] = l[r] * alpha + p0 + p1;
      o0[r] *= alpha;
      o1[r] *= alpha;
      pbuf[(rowb + r) * KP + colb]      = (_Float16)p0;
      pbuf[(rowb + r) * KP + 16 + colb] = (_Float16)p1;
      m[r] = mnew;
    }
    __syncthreads();

    const v16h pa  = load_a_frag(pbuf, KP, 0, 0, lane);
    const v16h vb0 = load_b_frag_nk(kbufT, KP, 0, 0, lane);   // cols = hd 0..15
    const v16h vb1 = load_b_frag_nk(kbufT, KP, 16, 0, lane);  // cols = hd 16..31
    o0 = wmma_f16(pa, vb0, o0);
    o1 = wmma_f16(pa, vb1, o1);
    __syncthreads();
  }

#pragma unroll
  for (int r = 0; r < 8; ++r) {
    float t = l[r];
    t += __shfl_xor(t, 1, 32);
    t += __shfl_xor(t, 2, 32);
    t += __shfl_xor(t, 4, 32);
    t += __shfl_xor(t, 8, 32);
    const float inv  = 1.0f / t;
    const int   row  = q0 + rowb + r;
    const size_t base = ((size_t)b * NN + row) * DMODEL + h * HDD;
    x1[base + colb]      = x[base + colb]      + o0[r] * inv;
    x1[base + 16 + colb] = x[base + 16 + colb] + o1[r] * inv;
  }
}

// ---------------- MLP GEMM1: h = gelu(y @ W1 + b1)  (exact erf GELU) --------
__global__ __launch_bounds__(32) void gemm1_gelu(const _Float16* __restrict__ yh,
                                                 const _Float16* __restrict__ w1t,
                                                 const float* __restrict__ b1,
                                                 _Float16* __restrict__ hh) {
  const int lane = threadIdx.x;
  const int n0 = blockIdx.x * 16;
  const int m0 = blockIdx.y * 16;
  v8f c = {};
#pragma unroll
  for (int k = 0; k < DMODEL; k += 32) {
    const v16h a  = load_a_frag(yh, DMODEL, m0, k, lane);
    const v16h bb = load_b_frag_nk(w1t, DMODEL, n0, k, lane);
    c = wmma_f16(a, bb, c);
  }
  const int   col  = n0 + (lane & 15);
  const int   rowb = (lane >> 4) * 8;
  const float bias = b1[col];
#pragma unroll
  for (int r = 0; r < 8; ++r) {
    const float vv = c[r] + bias;
    const float g  = 0.5f * vv * (1.0f + erff(vv * 0.70710678118654752f));
    hh[(size_t)(m0 + rowb + r) * DFF + col] = (_Float16)g;
  }
}

// ---------------- MLP GEMM2 + residual: out = x1 + h @ W2 + b2 ---------------
__global__ __launch_bounds__(32) void gemm2_out(const _Float16* __restrict__ hh,
                                                const _Float16* __restrict__ w2t,
                                                const float* __restrict__ b2,
                                                const float* __restrict__ x1,
                                                float* __restrict__ out) {
  const int lane = threadIdx.x;
  const int n0 = blockIdx.x * 16;
  const int m0 = blockIdx.y * 16;
  v8f c = {};
#pragma unroll 4
  for (int k = 0; k < DFF; k += 32) {
    const v16h a  = load_a_frag(hh, DFF, m0, k, lane);
    const v16h bb = load_b_frag_nk(w2t, DFF, n0, k, lane);
    c = wmma_f16(a, bb, c);
  }
  const int   col  = n0 + (lane & 15);
  const int   rowb = (lane >> 4) * 8;
  const float bias = b2[col];
#pragma unroll
  for (int r = 0; r < 8; ++r) {
    const size_t idx = (size_t)(m0 + rowb + r) * DMODEL + col;
    out[idx] = x1[idx] + c[r] + bias;
  }
}

// ---------------- Weight transpose + f32->f16 convert ------------------------
__global__ void transpose_f32_to_f16(const float* __restrict__ src,
                                     _Float16* __restrict__ dst, int K, int Nn) {
  const int idx = blockIdx.x * blockDim.x + threadIdx.x;
  if (idx >= K * Nn) return;
  const int k = idx / Nn;
  const int n = idx % Nn;
  dst[(size_t)n * K + k] = (_Float16)src[idx];
}

extern "C" void kernel_launch(void* const* d_in, const int* in_sizes, int n_in,
                              void* d_out, int out_size, void* d_ws, size_t ws_size,
                              hipStream_t stream) {
  const float* x   = (const float*)d_in[0];
  const int*   adj = (const int*)d_in[1];
  const float* g1  = (const float*)d_in[2];
  const float* be1 = (const float*)d_in[3];
  const float* g2  = (const float*)d_in[4];
  const float* be2 = (const float*)d_in[5];
  const float* W1  = (const float*)d_in[6];
  const float* b1  = (const float*)d_in[7];
  const float* W2  = (const float*)d_in[8];
  const float* b2  = (const float*)d_in[9];
  float* out = (float*)d_out;

  char* ws = (char*)d_ws;
  size_t off = 0;
  _Float16* xn  = (_Float16*)(ws + off); off += (size_t)BB * NN * DMODEL * 2;  // 4 MB
  float*    x1  = (float*)(ws + off);    off += (size_t)BB * NN * DMODEL * 4;  // 8 MB
  _Float16* yh  = (_Float16*)(ws + off); off += (size_t)BB * NN * DMODEL * 2;  // 4 MB
  _Float16* hh  = (_Float16*)(ws + off); off += (size_t)BB * NN * DFF * 2;     // 16 MB
  _Float16* w1t = (_Float16*)(ws + off); off += (size_t)DMODEL * DFF * 2;      // 0.5 MB
  _Float16* w2t = (_Float16*)(ws + off); off += (size_t)DMODEL * DFF * 2;      // 0.5 MB

  const int wcount = DMODEL * DFF;
  transpose_f32_to_f16<<<(wcount + 255) / 256, 256, 0, stream>>>(W1, w1t, DMODEL, DFF);
  transpose_f32_to_f16<<<(wcount + 255) / 256, 256, 0, stream>>>(W2, w2t, DFF, DMODEL);

  ln_to_f16<<<BB * NN, 32, 0, stream>>>(x, g1, be1, xn);

  attn_kernel<<<dim3(NN / 16, HH, BB), 32, 0, stream>>>(x, adj, xn, x1);

  ln_to_f16<<<BB * NN, 32, 0, stream>>>(x1, g2, be2, yh);

  gemm1_gelu<<<dim3(DFF / 16, (BB * NN) / 16), 32, 0, stream>>>(yh, w1t, b1, hh);
  gemm2_out<<<dim3(DMODEL / 16, (BB * NN) / 16), 32, 0, stream>>>(hh, w2t, b2, x1, out);

  (void)in_sizes; (void)n_in; (void)out_size; (void)ws_size;
}
